// DummyModel_26594437497041
// MI455X (gfx1250) — matrix operand order repaired and checked by reading, not compiled, not run on hardware
//
#include <hip/hip_runtime.h>
#include <hip/hip_bf16.h>

// ---------------------------------------------------------------------------
// 2-layer LSTM LM on MI455X (gfx1250), wave32 + WMMA bf16.
//   S=128, B=32, V=32000, H=256, 4H=1024.
// ---------------------------------------------------------------------------

#define S_LEN 128
#define B_SZ  32
#define V_SZ  32000
#define H_SZ  256
#define G_SZ  1024   // 4H

typedef __attribute__((ext_vector_type(16))) __bf16 v16bf;
typedef __attribute__((ext_vector_type(8)))  __bf16 v8bf;
typedef __attribute__((ext_vector_type(8)))  float  v8f;

static __device__ __forceinline__ unsigned short f2bf(float f) {
    union { float f; unsigned u; } v; v.f = f;
    unsigned r = v.u + 0x7FFFu + ((v.u >> 16) & 1u);   // round-to-nearest-even
    return (unsigned short)(r >> 16);
}

static __device__ __forceinline__ float sigmoid_f(float x) {
    return 1.0f / (1.0f + __expf(-x));
}
static __device__ __forceinline__ float tanh_f(float x) {
    float e = __expf(-2.0f * x);
    return (1.0f - e) / (1.0f + e);
}

// A-fragment (16-bit A, 16x32): per lane 16 contiguous elements,
// K = base..base+15  (base = 0 for lanes 0-15, 16 for 16-31)
static __device__ __forceinline__ v16bf load_a16(const unsigned short* p) {
    v8bf lo = *reinterpret_cast<const v8bf*>(p);
    v8bf hi = *reinterpret_cast<const v8bf*>(p + 8);
    return __builtin_shufflevector(lo, hi, 0,1,2,3,4,5,6,7,8,9,10,11,12,13,14,15);
}

// ---------------------------------------------------------------------------
// f32 -> bf16 elementwise conversion
// ---------------------------------------------------------------------------
__global__ void cvt_bf16_kernel(const float* __restrict__ src,
                                unsigned short* __restrict__ dst, int n) {
    int i = blockIdx.x * 256 + threadIdx.x;
    if (i < n) dst[i] = f2bf(src[i]);
}

// ---------------------------------------------------------------------------
// xp1[s][b][g] = w_ih1[g][tok(s,b)] + b_ih1[g] + b_hh1[g]
// ---------------------------------------------------------------------------
__global__ void embed_kernel(const int* __restrict__ x,
                             const float* __restrict__ w_ih1,
                             const float* __restrict__ b_ih1,
                             const float* __restrict__ b_hh1,
                             float* __restrict__ xp) {
    int idx = blockIdx.x * 256 + threadIdx.x;     // < 4096*1024
    int col = idx & (G_SZ - 1);
    int row = idx >> 10;                          // s*B + b
    int tok = x[row];
    xp[idx] = w_ih1[(size_t)col * V_SZ + tok] + b_ih1[col] + b_hh1[col];
}

// ---------------------------------------------------------------------------
// One LSTM layer, whole sequence, single persistent 1024-thread workgroup.
//   xp   : (S,B,4H) f32 precomputed input projections (incl. biases)
//   whh  : (4H,H)  bf16 row-major (= columns of B matrix, K-contiguous)
//   hout : (S,B,H) bf16 hidden states (input to the next GEMM)
// Wave w owns output tile (mA = w>>4, jt = w&15) for all four gates; the
// c-state tile stays in that wave's registers across all 128 steps; h
// ping-pongs through LDS. w_hh is re-streamed from L2 every step: the opaque
// ZERO OFFSET below is redefined by volatile asm each iteration, so LICM
// cannot hoist the 256 VGPRs of B-fragments out of the loop (no spills), while
// the pointer stays a GEP off the kernarg so address-space inference keeps the
// loads on the GLOBAL path (global_load_b128, no flat/DScnt coupling).
// ---------------------------------------------------------------------------
__global__ __launch_bounds__(1024) void lstm_layer_kernel(
        const float* __restrict__ xp,
        const unsigned short* __restrict__ whh,
        unsigned short* __restrict__ hout) {
    __shared__ __align__(32) unsigned short hbuf[2][B_SZ * H_SZ];   // 2 x 16KB

    const int tid  = threadIdx.x;
    const int lane = tid & 31;
    const int wave = tid >> 5;
    const int mA   = wave >> 4;          // 0..1  (row tile)
    const int jt   = wave & 15;          // 0..15 (col tile within each gate)
    const int ln16 = lane & 15;
    const int hi16 = (lane >> 4) & 1;

    // h_{-1} = 0
    for (int i = tid; i < B_SZ * H_SZ; i += 1024) hbuf[0][i] = 0;
    __syncthreads();

    v8f c;
    #pragma unroll
    for (int r = 0; r < 8; ++r) c[r] = 0.0f;

    const int arow  = mA * 16 + ln16;    // A-fragment row (b)
    const int abase = hi16 * 16;         // A-fragment K base
    const int kbB   = hi16 * 16;         // B-fragment K base

    for (int s = 0; s < S_LEN; ++s) {
        const unsigned short* prev = hbuf[s & 1];
        unsigned short*       next = hbuf[(s + 1) & 1];

        // Opaque zero offset: keeps weight loads inside the loop but on the
        // global address-space path.
        size_t zoff = 0;
        asm volatile("" : "+s"(zoff));
        const unsigned short* whh_s = whh + zoff;

        // ---- A fragments of h_{s-1} from LDS (8 K-slices of 32) ----
        v16bf afrag[8];
        #pragma unroll
        for (int kk = 0; kk < 8; ++kk)
            afrag[kk] = load_a16(prev + arow * H_SZ + kk * 32 + abase);

        // ---- accumulators initialized with xp_t (gate order i,f,g,o) ----
        v8f acc[4];
        #pragma unroll
        for (int g = 0; g < 4; ++g) {
            const int col = g * H_SZ + jt * 16 + ln16;
            #pragma unroll
            for (int r = 0; r < 8; ++r) {
                const int b = mA * 16 + hi16 * 8 + r;
                acc[g][r] = xp[((size_t)s * B_SZ + b) * G_SZ + col];
            }
        }

        // ---- gates += h_{s-1} @ whh^T ----
        // kk-outer / gate-inner: 4 B-fragment loads issue together, then 4
        // WMMAs that all reuse afrag[kk].
        const unsigned short* wrow0 = whh_s + (size_t)(0 * H_SZ + jt * 16 + ln16) * H_SZ;
        const unsigned short* wrow1 = whh_s + (size_t)(1 * H_SZ + jt * 16 + ln16) * H_SZ;
        const unsigned short* wrow2 = whh_s + (size_t)(2 * H_SZ + jt * 16 + ln16) * H_SZ;
        const unsigned short* wrow3 = whh_s + (size_t)(3 * H_SZ + jt * 16 + ln16) * H_SZ;
        #pragma unroll
        for (int kk = 0; kk < 8; ++kk) {
            v16bf b0 = *reinterpret_cast<const v16bf*>(wrow0 + kk * 32 + kbB);
            v16bf b1 = *reinterpret_cast<const v16bf*>(wrow1 + kk * 32 + kbB);
            v16bf b2 = *reinterpret_cast<const v16bf*>(wrow2 + kk * 32 + kbB);
            v16bf b3 = *reinterpret_cast<const v16bf*>(wrow3 + kk * 32 + kbB);
            acc[0] = __builtin_amdgcn_wmma_f32_16x16x32_bf16(
                false, afrag[kk], false, b0, (short)0, acc[0], false, false);
            acc[1] = __builtin_amdgcn_wmma_f32_16x16x32_bf16(
                false, afrag[kk], false, b1, (short)0, acc[1], false, false);
            acc[2] = __builtin_amdgcn_wmma_f32_16x16x32_bf16(
                false, afrag[kk], false, b2, (short)0, acc[2], false, false);
            acc[3] = __builtin_amdgcn_wmma_f32_16x16x32_bf16(
                false, afrag[kk], false, b3, (short)0, acc[3], false, false);
        }

        // ---- pointwise LSTM cell; c stays resident in this wave ----
        #pragma unroll
        for (int r = 0; r < 8; ++r) {
            float iv = sigmoid_f(acc[0][r]);
            float fv = sigmoid_f(acc[1][r]);
            float gv = tanh_f   (acc[2][r]);
            float ov = sigmoid_f(acc[3][r]);
            float cv = fv * c[r] + iv * gv;
            c[r] = cv;
            float hv = ov * tanh_f(cv);
            unsigned short hb = f2bf(hv);
            const int b = mA * 16 + hi16 * 8 + r;
            const int j = jt * 16 + ln16;
            next[b * H_SZ + j] = hb;                               // for step s+1
            hout[((size_t)s * B_SZ + b) * H_SZ + j] = hb;          // for next GEMM
        }
        __syncthreads();
    }
}

// ---------------------------------------------------------------------------
// xp2 = h1 @ w_ih2^T + (b_ih2 + b_hh2)       (4096 x 256)·(256 x 1024)
// One 16x16 output tile per wave, 8 waves / block.
// ---------------------------------------------------------------------------
__global__ __launch_bounds__(256) void gemm_xp2_kernel(
        const unsigned short* __restrict__ a,    // (4096,256) bf16
        const unsigned short* __restrict__ wb,   // (1024,256) bf16
        const float* __restrict__ b1,
        const float* __restrict__ b2,
        float* __restrict__ out) {               // (4096,1024) f32
    const int lane = threadIdx.x & 31;
    const int wave = threadIdx.x >> 5;
    const int tile = blockIdx.x * 8 + wave;      // 0..16383
    const int mt   = tile >> 6;                  // 0..255
    const int nt   = tile & 63;                  // 0..63
    const int ln16 = lane & 15;
    const int hi16 = (lane >> 4) & 1;

    const int col = nt * 16 + ln16;
    const float bias = b1[col] + b2[col];
    v8f acc;
    #pragma unroll
    for (int r = 0; r < 8; ++r) acc[r] = bias;

    const unsigned short* arow = a  + (size_t)(mt * 16 + ln16) * H_SZ;
    const unsigned short* wrow = wb + (size_t)col * H_SZ;
    #pragma unroll
    for (int kk = 0; kk < 8; ++kk) {
        v16bf af = load_a16(arow + kk * 32 + hi16 * 16);
        v16bf bf = *reinterpret_cast<const v16bf*>(wrow + kk * 32 + hi16 * 16);
        acc = __builtin_amdgcn_wmma_f32_16x16x32_bf16(
            false, af, false, bf, (short)0, acc, false, false);
    }

    const int rowb = mt * 16 + hi16 * 8;
    #pragma unroll
    for (int r = 0; r < 8; ++r)
        out[(size_t)(rowb + r) * G_SZ + col] = acc[r];
}

// ---------------------------------------------------------------------------
// out = h2 @ w_out^T + b_out                  (4096 x 256)·(256 x 32000)
// Store-bandwidth bound (524 MB f32 out). Each wave: 1 M-tile x 4 N-tiles,
// A-fragments loaded once and reused across the 4 N-tiles.
// ---------------------------------------------------------------------------
__global__ __launch_bounds__(256) void gemm_out_kernel(
        const unsigned short* __restrict__ a,    // (4096,256) bf16
        const unsigned short* __restrict__ wb,   // (32000,256) bf16
        const float* __restrict__ bias,          // (32000)
        float* __restrict__ out) {               // (4096,32000) f32
    const int lane = threadIdx.x & 31;
    const int wave = threadIdx.x >> 5;
    const int g    = blockIdx.x * 8 + wave;      // 0..127999
    const int mt   = g / 500;                    // 0..255
    const int ng   = g % 500;                    // group of 4 N-tiles
    const int ln16 = lane & 15;
    const int hi16 = (lane >> 4) & 1;

    v16bf afrag[8];
    const unsigned short* arow = a + (size_t)(mt * 16 + ln16) * H_SZ;
    #pragma unroll
    for (int kk = 0; kk < 8; ++kk)
        afrag[kk] = load_a16(arow + kk * 32 + hi16 * 16);

    const int rowb = mt * 16 + hi16 * 8;
    #pragma unroll
    for (int t = 0; t < 4; ++t) {
        const int v = (ng * 4 + t) * 16 + ln16;              // vocab column
        const unsigned short* wrow = wb + (size_t)v * H_SZ;
        const float bv = bias[v];
        v8f acc;
        #pragma unroll
        for (int r = 0; r < 8; ++r) acc[r] = bv;
        #pragma unroll
        for (int kk = 0; kk < 8; ++kk) {
            v16bf bf = *reinterpret_cast<const v16bf*>(wrow + kk * 32 + hi16 * 16);
            acc = __builtin_amdgcn_wmma_f32_16x16x32_bf16(
                false, afrag[kk], false, bf, (short)0, acc, false, false);
        }
        #pragma unroll
        for (int r = 0; r < 8; ++r)
            out[(size_t)(rowb + r) * V_SZ + v] = acc[r];
    }
}

// ---------------------------------------------------------------------------
// Host-side launcher
// ---------------------------------------------------------------------------
extern "C" void kernel_launch(void* const* d_in, const int* in_sizes, int n_in,
                              void* d_out, int out_size, void* d_ws, size_t ws_size,
                              hipStream_t stream) {
    // setup_inputs order:
    // 0:x 1:w_ih1 2:w_hh1 3:b_ih1 4:b_hh1 5:w_ih2 6:w_hh2 7:b_ih2 8:b_hh2 9:w_out 10:b_out
    const int*   x      = (const int*)  d_in[0];
    const float* w_ih1  = (const float*)d_in[1];
    const float* w_hh1  = (const float*)d_in[2];
    const float* b_ih1  = (const float*)d_in[3];
    const float* b_hh1  = (const float*)d_in[4];
    const float* w_ih2  = (const float*)d_in[5];
    const float* w_hh2  = (const float*)d_in[6];
    const float* b_ih2  = (const float*)d_in[7];
    const float* b_hh2  = (const float*)d_in[8];
    const float* w_out  = (const float*)d_in[9];
    const float* b_out  = (const float*)d_in[10];
    float* out = (float*)d_out;

    char* ws = (char*)d_ws;
    // Workspace layout (bytes); xp buffer reused for xp1 then xp2.
    float*          xp      = (float*)         (ws + 0);          // 16,777,216 B
    unsigned short* h1_bf   = (unsigned short*)(ws + 16777216);   //  2,097,152 B
    unsigned short* h2_bf   = (unsigned short*)(ws + 18874368);   //  2,097,152 B
    unsigned short* whh1_bf = (unsigned short*)(ws + 20971520);   //    524,288 B
    unsigned short* whh2_bf = (unsigned short*)(ws + 21495808);   //    524,288 B
    unsigned short* wih2_bf = (unsigned short*)(ws + 22020096);   //    524,288 B
    unsigned short* wout_bf = (unsigned short*)(ws + 22544384);   // 16,384,000 B
    (void)ws_size; (void)in_sizes; (void)n_in; (void)out_size;

    // Weight conversions (independent; L2-resident afterwards)
    cvt_bf16_kernel<<<(G_SZ * H_SZ + 255) / 256, 256, 0, stream>>>(w_hh1, whh1_bf, G_SZ * H_SZ);
    cvt_bf16_kernel<<<(G_SZ * H_SZ + 255) / 256, 256, 0, stream>>>(w_hh2, whh2_bf, G_SZ * H_SZ);
    cvt_bf16_kernel<<<(G_SZ * H_SZ + 255) / 256, 256, 0, stream>>>(w_ih2, wih2_bf, G_SZ * H_SZ);
    cvt_bf16_kernel<<<(V_SZ * H_SZ + 255) / 256, 256, 0, stream>>>(w_out, wout_bf, V_SZ * H_SZ);

    // xp1 = gather(w_ih1, x) + b_ih1 + b_hh1
    embed_kernel<<<(S_LEN * B_SZ * G_SZ) / 256, 256, 0, stream>>>(x, w_ih1, b_ih1, b_hh1, xp);

    // LSTM layer 1 (persistent single workgroup, 32 waves)
    lstm_layer_kernel<<<1, 1024, 0, stream>>>(xp, whh1_bf, h1_bf);

    // xp2 = h1 @ w_ih2^T + biases   (reuses xp buffer)
    gemm_xp2_kernel<<<2048, 256, 0, stream>>>(h1_bf, wih2_bf, b_ih2, b_hh2, xp);

    // LSTM layer 2
    lstm_layer_kernel<<<1, 1024, 0, stream>>>(xp, whh2_bf, h2_bf);

    // logits = h2 @ w_out^T + b_out
    gemm_out_kernel<<<16000, 256, 0, stream>>>(h2_bf, wout_bf, b_out, out);
}
